// Quadratic_40200893891171
// MI455X (gfx1250) — matrix operand order repaired and checked by reading, not compiled, not run on hardware
//
#include <hip/hip_runtime.h>
#include <stdint.h>

typedef __bf16 bf16;
typedef __attribute__((ext_vector_type(16))) __bf16 v16bf;
typedef __attribute__((ext_vector_type(8)))  __bf16 v8bf;
typedef __attribute__((ext_vector_type(8)))  float  v8f;
typedef __attribute__((ext_vector_type(4)))  int    v4i;

#define D_DIM 1024
#define MTILE 64
#define KC    32     // K per staging phase (1 WMMA k-step), double-buffered
#define AROW  1032   // D + 8 pad  -> row stride 2064 B (16B aligned, +4 banks/row)
#define BROW  40     // KC + 8 pad -> row stride 80 B  (16B aligned, +20 banks/row)

// LDS layout (bytes)
#define AHI_OFF   0
#define ALO_OFF   (MTILE * AROW * 2)                    // 132096
#define B_HALF_B  (8 * 16 * BROW * 2)                   // 10240: one array (hi or lo), one buffer (8 tiles)
#define B_BUF_B   (2 * B_HALF_B)                        // 20480: hi+lo, one buffer
#define BBUF_OFF  (2 * MTILE * AROW * 2)                // 264192
#define YRED_OFF  (BBUF_OFF + 2 * B_BUF_B)              // 305152
#define LDS_BYTES (YRED_OFF + MTILE * 4)                // 305408  (< 320KB WGP LDS)

// ---- CDNA5 async global->LDS copy (ASYNCcnt) ----
typedef __attribute__((address_space(1))) v4i* as1_v4i_p;
typedef __attribute__((address_space(3))) v4i* as3_v4i_p;

#if __has_builtin(__builtin_amdgcn_global_load_async_to_lds_b128)
#define USE_ASYNC_LDS 1
__device__ __forceinline__ void async_cp16(const bf16* g, bf16* l) {
    __builtin_amdgcn_global_load_async_to_lds_b128(
        (as1_v4i_p)(void*)(const_cast<bf16*>(g)),
        (as3_v4i_p)(void*)l, 0, 0);
}
#else
#define USE_ASYNC_LDS 0
__device__ __forceinline__ void async_cp16(const bf16* g, bf16* l) {
    *(uint4*)l = *(const uint4*)g;     // synchronous fallback
}
#endif

#if __has_builtin(__builtin_amdgcn_s_wait_asynccnt)
#define WAIT_ASYNC0() __builtin_amdgcn_s_wait_asynccnt(0)
#else
#if USE_ASYNC_LDS
#define WAIT_ASYNC0() asm volatile("s_wait_asynccnt 0x0" ::: "memory")
#else
#define WAIT_ASYNC0() ((void)0)
#endif
#endif

__device__ __forceinline__ void split_bits(float x, unsigned short& h, unsigned short& l) {
    uint32_t u = __builtin_bit_cast(uint32_t, x);
    h = (unsigned short)(u >> 16);                       // truncated bf16 (hi)
    float hf = __builtin_bit_cast(float, (uint32_t)h << 16);
    bf16 lb = (bf16)(x - hf);                            // RNE bf16 of residual
    l = __builtin_bit_cast(unsigned short, lb);
}

// ---------------------------------------------------------------------------
// Kernel 1: split Q (fp32 [k][n]) into transposed bf16 hi/lo arrays Qt[n][k].
// ---------------------------------------------------------------------------
__global__ __launch_bounds__(256) void split_q_kernel(
    const float* __restrict__ Q, bf16* __restrict__ Qt_hi, bf16* __restrict__ Qt_lo) {
    __shared__ float tile[64][65];
    const int bk = blockIdx.x * 64;
    const int bn = blockIdx.y * 64;
    const int t = threadIdx.x;
    #pragma unroll
    for (int i = 0; i < 16; ++i) {
        int idx = t + 256 * i;
        int r = idx >> 6, cc = idx & 63;
        tile[r][cc] = Q[(size_t)(bk + r) * D_DIM + bn + cc];
    }
    __syncthreads();
    #pragma unroll
    for (int i = 0; i < 16; ++i) {
        int idx = t + 256 * i;
        int n = idx >> 6, k = idx & 63;
        unsigned short h, l;
        split_bits(tile[k][n], h, l);
        size_t o = (size_t)(bn + n) * D_DIM + bk + k;
        ((unsigned short*)Qt_hi)[o] = h;
        ((unsigned short*)Qt_lo)[o] = l;
    }
}

// ---------------------------------------------------------------------------
// Stage one KC=32 chunk (16 KB) of B into LDS buffer p: 1024 16B units,
// 4 per thread, issued as async global->LDS copies (no VGPR data path).
// unit u -> array (hi/lo), tile slot tq (quarter*2 + j), row nn, 16B offset.
// Column tile for slot tq at iteration ntd:  (tq>>1)*16 + ntd*2 + (tq&1).
// ---------------------------------------------------------------------------
__device__ __forceinline__ void stage_issue(int t, int ntd, int kc, char* smem, int p,
                                            const bf16* __restrict__ Qt_hi,
                                            const bf16* __restrict__ Qt_lo) {
    #pragma unroll
    for (int i = 0; i < 4; ++i) {
        int u = t + 256 * i;
        int a = u >> 9, rem = u & 511;
        int row = rem >> 2, off8 = (rem & 3) << 3;     // row 0..127, 16B unit in 32-elem row
        int tq = row >> 4, nn = row & 15;              // tile slot 0..7
        int quarter = tq >> 1, j = tq & 1;
        int ng = ((quarter * 16 + ntd * 2 + j) * 16 + nn);         // global column
        const bf16* src = (a ? Qt_lo : Qt_hi) + (size_t)ng * D_DIM + kc * KC + off8;
        bf16* dst = (bf16*)(smem + BBUF_OFF + p * B_BUF_B + a * B_HALF_B) + (size_t)row * BROW + off8;
        async_cp16(src, dst);
    }
}

// ---------------------------------------------------------------------------
// Kernel 2: fused  y = rowsum((x@Q + b) .* x) + c  via bf16x2-split WMMA.
// 8 waves: rgPair = w&1 -> rows rgPair*32..+31 (two 16-row tiles),
//          nQuarter = w>>1 -> 16 column tiles, processed 2 at a time (8 iters).
// Register tile M=32 x N=32: A loads shared across 2 n-tiles, B loads shared
// across 2 row-groups -> 16 ds_load_b128 per 12 WMMAs (operand-reuse floor).
// B chunks double-buffered, staged with GLOBAL_LOAD_ASYNC_TO_LDS_B128.
// ---------------------------------------------------------------------------
__global__ __launch_bounds__(256, 1) void quad_wmma_kernel(
    const float* __restrict__ x, const bf16* __restrict__ Qt_hi,
    const bf16* __restrict__ Qt_lo, const float* __restrict__ bvec,
    const float* __restrict__ cvec, float* __restrict__ out) {
    extern __shared__ char smem[];
    bf16*  Ahi  = (bf16*)(smem + AHI_OFF);    // [64][AROW]
    bf16*  Alo  = (bf16*)(smem + ALO_OFF);
    float* yred = (float*)(smem + YRED_OFF);  // [64]

    const int t    = threadIdx.x;
    const int lane = t & 31;
    const int wave = t >> 5;
    const int rgP  = wave & 1;       // row pair: rows rgP*32 .. +31
    const int nq   = wave >> 1;      // column quarter (16 tiles)
    const int lh   = lane >> 4;      // lane half (0/1)
    const int ln   = lane & 15;
    const int m0   = blockIdx.x * MTILE;
    const int rowBase = rgP * 32;

    if (t < MTILE) yred[t] = 0.0f;

    // ---- Stage x tile into LDS with hi/lo split ----
    for (int i = 0; i < 64; ++i) {
        int u  = t + 256 * i;
        int r  = u >> 8;
        int c4 = u & 255;
        const float4 v = *(const float4*)(x + (size_t)(m0 + r) * D_DIM + c4 * 4);
        unsigned short h0, l0, h1, l1, h2, l2, h3, l3;
        split_bits(v.x, h0, l0); split_bits(v.y, h1, l1);
        split_bits(v.z, h2, l2); split_bits(v.w, h3, l3);
        size_t eo = (size_t)r * AROW + c4 * 4;
        *(ushort4*)((unsigned short*)Ahi + eo) = make_ushort4(h0, h1, h2, h3);
        *(ushort4*)((unsigned short*)Alo + eo) = make_ushort4(l0, l1, l2, l3);
    }
    __syncthreads();

    v8f acc[2];
    acc[0] = (v8f){0.f,0.f,0.f,0.f,0.f,0.f,0.f,0.f};
    acc[1] = (v8f){0.f,0.f,0.f,0.f,0.f,0.f,0.f,0.f};

    for (int ntd = 0; ntd < 8; ++ntd) {          // 2 column tiles per iteration
        v8f C[4];                                // C[rr*2 + j]
        #pragma unroll
        for (int q = 0; q < 4; ++q) C[q] = (v8f){0.f,0.f,0.f,0.f,0.f,0.f,0.f,0.f};

        stage_issue(t, ntd, 0, smem, 0, Qt_hi, Qt_lo);   // prologue: chunk 0 -> buf 0

        for (int kc = 0; kc < 32; ++kc) {
            WAIT_ASYNC0();                       // my chunk-kc LDS writes complete
            __syncthreads();                     // everyone's writes visible; prev readers done
            if (kc < 31)                         // stage next chunk into other buffer
                stage_issue(t, ntd, kc + 1, smem, (kc + 1) & 1, Qt_hi, Qt_lo);

            // ---- compute chunk kc from buf[kc&1] ----
            const int p = kc & 1;
            const bf16* Bhi = (const bf16*)(smem + BBUF_OFF + p * B_BUF_B);
            const bf16* Blo = (const bf16*)(smem + BBUF_OFF + p * B_BUF_B + B_HALF_B);
            const int k0 = kc * KC;

            v16bf Ah[2], Al[2];
            #pragma unroll
            for (int rr = 0; rr < 2; ++rr) {
                const bf16* ap  = Ahi + (size_t)(rowBase + rr * 16 + ln) * AROW + k0 + 8 * lh;
                const bf16* alp = Alo + (size_t)(rowBase + rr * 16 + ln) * AROW + k0 + 8 * lh;
                v8bf a0  = *(const v8bf*)ap,  a1  = *(const v8bf*)(ap + 16);
                v8bf al0 = *(const v8bf*)alp, al1 = *(const v8bf*)(alp + 16);
                Ah[rr] = __builtin_shufflevector(a0,  a1,  0,1,2,3,4,5,6,7,8,9,10,11,12,13,14,15);
                Al[rr] = __builtin_shufflevector(al0, al1, 0,1,2,3,4,5,6,7,8,9,10,11,12,13,14,15);
            }
            #pragma unroll
            for (int j = 0; j < 2; ++j) {
                const int be = ((nq * 2 + j) * 16 + ln) * BROW + 8 * lh;
                const bf16* bp  = Bhi + be;
                const bf16* blp = Blo + be;
                v8bf b0  = *(const v8bf*)bp,  b1  = *(const v8bf*)(bp + 16);
                v8bf bl0 = *(const v8bf*)blp, bl1 = *(const v8bf*)(blp + 16);
                v16bf Bh = __builtin_shufflevector(b0,  b1,  0,1,2,3,4,5,6,7,8,9,10,11,12,13,14,15);
                v16bf Bl = __builtin_shufflevector(bl0, bl1, 0,1,2,3,4,5,6,7,8,9,10,11,12,13,14,15);
                #pragma unroll
                for (int rr = 0; rr < 2; ++rr) {
                    C[rr*2+j] = __builtin_amdgcn_wmma_f32_16x16x32_bf16(false, Ah[rr], false, Bh, (short)0, C[rr*2+j], false, false);
                    C[rr*2+j] = __builtin_amdgcn_wmma_f32_16x16x32_bf16(false, Ah[rr], false, Bl, (short)0, C[rr*2+j], false, false);
                    C[rr*2+j] = __builtin_amdgcn_wmma_f32_16x16x32_bf16(false, Al[rr], false, Bh, (short)0, C[rr*2+j], false, false);
                }
            }
        }

        // ---- Fused reduction: acc[rr][v] += (C[rr*2+j][v] + b_n) * x[m, n] ----
        #pragma unroll
        for (int j = 0; j < 2; ++j) {
            const int ng = (nq * 16 + ntd * 2 + j) * 16 + ln;     // global column
            const float bn = bvec[ng];
            #pragma unroll
            for (int rr = 0; rr < 2; ++rr) {
                #pragma unroll
                for (int v = 0; v < 8; ++v) {
                    int ml = rowBase + rr * 16 + v + 8 * lh;      // C layout: M = v + 8*laneHalf
                    float xv = (float)Ahi[(size_t)ml * AROW + ng] + (float)Alo[(size_t)ml * AROW + ng];
                    acc[rr][v] += (C[rr*2+j][v] + bn) * xv;
                }
            }
        }
    }

    // ---- Cross-lane reduce over the 16 columns held per lane half ----
    #pragma unroll
    for (int rr = 0; rr < 2; ++rr) {
        #pragma unroll
        for (int v = 0; v < 8; ++v) {
            float s = acc[rr][v];
            s += __shfl_xor(s, 1);
            s += __shfl_xor(s, 2);
            s += __shfl_xor(s, 4);
            s += __shfl_xor(s, 8);
            if (ln == 0) atomicAdd(&yred[rowBase + rr * 16 + v + 8 * lh], s);  // ds_add_f32
        }
    }
    __syncthreads();
    if (t < MTILE) out[m0 + t] = yred[t] + cvec[0];
}

extern "C" void kernel_launch(void* const* d_in, const int* in_sizes, int n_in,
                              void* d_out, int out_size, void* d_ws, size_t ws_size,
                              hipStream_t stream) {
    const float* x = (const float*)d_in[0];
    const float* Q = (const float*)d_in[1];
    const float* b = (const float*)d_in[2];
    const float* c = (const float*)d_in[3];
    float* out = (float*)d_out;

    bf16* Qt_hi = (bf16*)d_ws;
    bf16* Qt_lo = Qt_hi + (size_t)D_DIM * D_DIM;

    (void)hipFuncSetAttribute((const void*)quad_wmma_kernel,
                              hipFuncAttributeMaxDynamicSharedMemorySize, LDS_BYTES);

    dim3 g1(D_DIM / 64, D_DIM / 64);
    split_q_kernel<<<g1, 256, 0, stream>>>(Q, Qt_hi, Qt_lo);

    const int nblocks = 16384 / MTILE;   // 256
    quad_wmma_kernel<<<nblocks, 256, LDS_BYTES, stream>>>(x, Qt_hi, Qt_lo, b, c, out);
}
